// ConvTreeBlock_14474039787888
// MI455X (gfx1250) — compile-verified, matrix operand order checked by reading, not run on hardware
//
#include <hip/hip_runtime.h>

#define NPTS   1000000
#define KSZ    9
#define CH     16
#define KDIM   144            // KSZ*CH
#define NCHUNK 5              // ceil(144/32) -> K padded to 160
#define ASTR   168            // LDS row stride in halves (bank-conflict padding)
#define WPB    8              // waves per block
#define EPSV   1e-5f
#define SLOPE  0.2f

typedef _Float16 h4   __attribute__((ext_vector_type(4)));
typedef _Float16 h8   __attribute__((ext_vector_type(8)));
typedef _Float16 v16h __attribute__((ext_vector_type(16)));
typedef float    v8f  __attribute__((ext_vector_type(8)));

__device__ __forceinline__ float leaky(float x) { return x >= 0.f ? x : SLOPE * x; }

// ---------------------------------------------------------------------------
// Pack w1/w2 into per-lane WMMA B fragments (K padded to 160 with zeros),
// fragment-element order so the GEMM kernel loads each lane's 16 halves
// contiguously. Also zero the 64 global stat accumulators (every call).
// B[k][o] = w[c][o][j], k = j*16 + c. Fragment layout (16-bit B, 32x16):
// lane L: o = L&15, kbase = (L>>4)*8; e<8 -> K=kbase+e, e>=8 -> K=kbase+8+e.
// ---------------------------------------------------------------------------
__global__ void prep_kernel(const float* __restrict__ w1, const float* __restrict__ w2,
                            _Float16* __restrict__ bpk1, _Float16* __restrict__ bpk2,
                            float* __restrict__ stats /* 64 floats */) {
  int tid = threadIdx.x;
  if (tid < 320) {
    const float* w  = (tid < 160) ? w1 : w2;
    _Float16*    bp = (tid < 160) ? bpk1 : bpk2;
    int f = (tid < 160) ? tid : tid - 160;     // fragment id: t*32 + lane
    int t = f >> 5, lane = f & 31;
    int o = lane & 15, kb = (lane >> 4) * 8;
#pragma unroll
    for (int e = 0; e < 16; ++e) {
      int k = t * 32 + kb + (e < 8 ? e : 8 + e);
      float v = 0.f;
      if (k < KDIM) {
        int j = k >> 4, c = k & 15;
        v = w[(c * CH + o) * KSZ + j];
      }
      bp[f * 16 + e] = (_Float16)v;
    }
  } else if (tid < 384) {
    stats[tid - 320] = 0.f;
  }
}

// ---------------------------------------------------------------------------
// One 16-row output tile per wave: gather 144 rows into an LDS A-tile,
// (optionally fused BN-affine + leaky on the gathered values), then
// 5x v_wmma_f32_16x16x32_f16 with f32 accumulation (bias in accumulator).
// Also accumulates per-channel sum / sum-of-squares for batch stats.
// ---------------------------------------------------------------------------
__global__ void __launch_bounds__(256) treeconv_gemm(
    const float* __restrict__ src, const int* __restrict__ ind,
    const _Float16* __restrict__ bpk,
    const float* __restrict__ bias,    // nullable (pass 1)
    const float* __restrict__ aff_s,   // nullable (pass 2: fused BN+leaky)
    const float* __restrict__ aff_t,
    float* __restrict__ dst,
    float* __restrict__ gsum, float* __restrict__ gsq) {
  __shared__ _Float16 Atile[WPB][CH][ASTR];
  __shared__ float ssum[CH], ssq[CH];

  const int tid  = threadIdx.x;
  const int wave = tid >> 5, lane = tid & 31;
  const int tileBase = (blockIdx.x * WPB + wave) * 16;
  const bool active  = tileBase < NPTS;

  if (tid < 32) {
    if (tid < 16) ssum[tid] = 0.f; else ssq[tid - 16] = 0.f;
  }

  if (active) {
    // zero the K-padding region [144,160) of this wave's tile
    {
      int m = lane >> 1;
      int off = KDIM + (lane & 1) * 8;
      h8 z;
#pragma unroll
      for (int i = 0; i < 8; ++i) z[i] = (_Float16)0.f;
      *(h8*)&Atile[wave][m][off] = z;
    }
    float4 S[4], T[4];
    if (aff_s) {
#pragma unroll
      for (int q = 0; q < 4; ++q) {
        S[q] = ((const float4*)aff_s)[q];
        T[q] = ((const float4*)aff_t)[q];
      }
    }
    // gather: 144 (m,j) rows of 16 floats, convert to f16 into Atile[m][j*16..]
    for (int p = lane; p < KDIM; p += 32) {
      int m = p / KSZ, j = p - m * KSZ;
      int row = ind[(tileBase + m) * KSZ + j];
      const float4* r4 = (const float4*)(src + (long)row * CH);
#pragma unroll
      for (int q = 0; q < 4; ++q) {
        float4 v = r4[q];
        if (aff_s) {
          v.x = leaky(fmaf(S[q].x, v.x, T[q].x));
          v.y = leaky(fmaf(S[q].y, v.y, T[q].y));
          v.z = leaky(fmaf(S[q].z, v.z, T[q].z));
          v.w = leaky(fmaf(S[q].w, v.w, T[q].w));
        }
        h4 h;
        h[0] = (_Float16)v.x; h[1] = (_Float16)v.y;
        h[2] = (_Float16)v.z; h[3] = (_Float16)v.w;
        *(h4*)&Atile[wave][m][j * 16 + q * 4] = h;
      }
    }
  }
  __syncthreads();

  if (active) {
    const int o  = lane & 15;       // output channel (also A row: m = lane&15)
    const int hi = lane >> 4;       // lane-half selector
    v8f acc;
    const float binit = bias ? bias[o] : 0.f;
#pragma unroll
    for (int r = 0; r < 8; ++r) acc[r] = binit;

#pragma unroll
    for (int t = 0; t < NCHUNK; ++t) {
      const int k0 = t * 32 + hi * 8;
      h8 alo = *(const h8*)&Atile[wave][o][k0];
      h8 ahi = *(const h8*)&Atile[wave][o][k0 + 16];
      v16h a;
#pragma unroll
      for (int i = 0; i < 8; ++i) { a[i] = alo[i]; a[i + 8] = ahi[i]; }
      v16h b = *(const v16h*)(bpk + ((t * 32 + lane) << 4));
      acc = __builtin_amdgcn_wmma_f32_16x16x32_f16(
          false, a, false, b, (short)0, acc, false, false);
    }

    float psum = 0.f, psq = 0.f;
#pragma unroll
    for (int r = 0; r < 8; ++r) {
      float v = acc[r];
      dst[(long)(tileBase + hi * 8 + r) * CH + o] = v;   // 64B-coalesced per half-wave
      psum += v;
      psq  += v * v;
    }
    atomicAdd(&ssum[o], psum);   // ds_add_f32
    atomicAdd(&ssq[o],  psq);
  }
  __syncthreads();

  if (tid < 16) {
    unsafeAtomicAdd(&gsum[tid], ssum[tid]);
    unsafeAtomicAdd(&gsq[tid],  ssq[tid]);
  }
}

// Per-channel: s = gamma * rsqrt(var + eps), t = beta - mean*s (biased var).
__global__ void finalize_stats(const float* __restrict__ gsum, const float* __restrict__ gsq,
                               const float* __restrict__ gamma, const float* __restrict__ beta,
                               float* __restrict__ s, float* __restrict__ t) {
  int i = threadIdx.x;
  if (i < CH) {
    const float invN = 1.f / (float)NPTS;
    float mean = gsum[i] * invN;
    float var  = gsq[i] * invN - mean * mean;
    float inv  = rsqrtf(var + EPSV);
    float sc   = gamma[i] * inv;
    s[i] = sc;
    t[i] = beta[i] - mean * sc;
  }
}

// out = leaky(s2[c]*x2 + t2[c] + data), float4-vectorized.
__global__ void final_fuse(const float* __restrict__ x2, const float* __restrict__ data,
                           const float* __restrict__ s2, const float* __restrict__ t2,
                           float* __restrict__ out) {
  int i = blockIdx.x * blockDim.x + threadIdx.x;   // float4 index
  if (i < NPTS * CH / 4) {
    float4 x = ((const float4*)x2)[i];
    float4 d = ((const float4*)data)[i];
    float4 sv = ((const float4*)s2)[i & 3];
    float4 tv = ((const float4*)t2)[i & 3];
    float4 o;
    o.x = leaky(fmaf(sv.x, x.x, tv.x) + d.x);
    o.y = leaky(fmaf(sv.y, x.y, tv.y) + d.y);
    o.z = leaky(fmaf(sv.z, x.z, tv.z) + d.z);
    o.w = leaky(fmaf(sv.w, x.w, tv.w) + d.w);
    ((float4*)out)[i] = o;
  }
}

extern "C" void kernel_launch(void* const* d_in, const int* in_sizes, int n_in,
                              void* d_out, int out_size, void* d_ws, size_t ws_size,
                              hipStream_t stream) {
  const float* data   = (const float*)d_in[0];
  const int*   ind    = (const int*)d_in[1];
  const float* w1     = (const float*)d_in[2];
  const float* b1     = (const float*)d_in[3];
  const float* gamma1 = (const float*)d_in[4];
  const float* beta1  = (const float*)d_in[5];
  const float* w2     = (const float*)d_in[6];
  const float* gamma2 = (const float*)d_in[7];
  const float* beta2  = (const float*)d_in[8];
  float* out = (float*)d_out;

  char* ws = (char*)d_ws;
  float*    x1   = (float*)ws;                                   // 64 MB
  float*    x2   = (float*)(ws + (size_t)NPTS * CH * 4);         // 64 MB
  _Float16* bpk1 = (_Float16*)(ws + (size_t)2 * NPTS * CH * 4);  // 5*32*16 halves
  _Float16* bpk2 = bpk1 + NCHUNK * 32 * 16;
  float* stats = (float*)(bpk2 + NCHUNK * 32 * 16);              // 128 floats
  float *gsum1 = stats,      *gsq1 = stats + 16;
  float *gsum2 = stats + 32, *gsq2 = stats + 48;
  float *s1 = stats + 64, *t1 = stats + 80;
  float *s2 = stats + 96, *t2 = stats + 112;

  prep_kernel<<<1, 384, 0, stream>>>(w1, w2, bpk1, bpk2, stats);

  const int tiles  = (NPTS + 15) / 16;                 // 62500
  const int blocks = (tiles + WPB - 1) / WPB;          // 7813

  treeconv_gemm<<<blocks, 256, 0, stream>>>(data, ind, bpk1, b1, nullptr, nullptr,
                                            x1, gsum1, gsq1);
  finalize_stats<<<1, 16, 0, stream>>>(gsum1, gsq1, gamma1, beta1, s1, t1);
  treeconv_gemm<<<blocks, 256, 0, stream>>>(x1, ind, bpk2, nullptr, s1, t1,
                                            x2, gsum2, gsq2);
  finalize_stats<<<1, 16, 0, stream>>>(gsum2, gsq2, gamma2, beta2, s2, t2);
  final_fuse<<<(NPTS * CH / 4 + 255) / 256, 256, 0, stream>>>(x2, data, s2, t2, out);
}
